// ProductionFastMQA_77687368450659
// MI455X (gfx1250) — compile-verified, matrix operand order checked by reading, not run on hardware
//
#include <hip/hip_runtime.h>
#include <hip/hip_bf16.h>

// ---------------------------------------------------------------------------
// MQA forward for MI455X (gfx1250): bf16 WMMA everywhere, flash-style softmax.
// Ones-column WMMA for the softmax denominator, Bc=64 key chunks, exp2-domain
// softmax with scale folded into the Q projection, async global->LDS staging
// in the GEMM (gfx1250 GLOBAL_LOAD_ASYNC_TO_LDS_B128 + s_wait_asynccnt).
// ---------------------------------------------------------------------------

typedef __attribute__((ext_vector_type(16))) __bf16 v16bf;
typedef __attribute__((ext_vector_type(8)))  float  v8f;

#define HIDDEN   2048
#define HEADS    16
#define HEAD_DIM 128
#define BATCH    2
#define SEQ      2048
#define NTOK     (BATCH * SEQ)   // 4096
#define ATTN_SCALE 0.08838834764831845f   // 1/sqrt(128)
#define LOG2E      1.4426950408889634f

#if defined(__HIPCC__) && defined(__has_builtin)
#if __has_builtin(__builtin_amdgcn_global_load_async_to_lds_b128) && \
    __has_builtin(__builtin_amdgcn_s_wait_asynccnt)
#define HAVE_ASYNC_LDS 1
#endif
#endif
#ifndef HAVE_ASYNC_LDS
#define HAVE_ASYNC_LDS 0
#endif

__device__ __forceinline__ float fast_exp2(float x) {
#if defined(__has_builtin) && __has_builtin(__builtin_amdgcn_exp2f)
  return __builtin_amdgcn_exp2f(x);
#else
  return exp2f(x);
#endif
}

__device__ __forceinline__ float fast_rcp(float x) {
#if defined(__has_builtin) && __has_builtin(__builtin_amdgcn_rcpf)
  return __builtin_amdgcn_rcpf(x);
#else
  return 1.0f / x;
#endif
}

// Stage 16 bytes global -> LDS; async DMA path on gfx1250 when available.
// Builtin signature (from hipcc diagnostic): (v4i AS1*, v4i AS3*, imm, imm).
__device__ __forceinline__ void stage16(const __bf16* __restrict__ g,
                                        __bf16* l) {
#if HAVE_ASYNC_LDS
  typedef int v4i __attribute__((vector_size(16)));
  typedef v4i __attribute__((address_space(1)))* gv4i_p;
  typedef v4i __attribute__((address_space(3)))* lv4i_p;
  __builtin_amdgcn_global_load_async_to_lds_b128((gv4i_p)(uintptr_t)g,
                                                 (lv4i_p)l, 0, 0);
#else
  *(uint4*)l = *(const uint4*)g;
#endif
}

__device__ __forceinline__ void stage_wait() {
#if HAVE_ASYNC_LDS
  __builtin_amdgcn_s_wait_asynccnt(0);
#endif
}

// ---------------------------------------------------------------------------
// Stage 0: fp32 -> bf16 conversion, 8 elements / thread iteration
// ---------------------------------------------------------------------------
__global__ void cvt_f32_bf16(const float* __restrict__ src,
                             __bf16* __restrict__ dst, int n8) {
  int i = blockIdx.x * blockDim.x + threadIdx.x;
  int stride = gridDim.x * blockDim.x;
  for (; i < n8; i += stride) {
    const float4* s = (const float4*)src + (size_t)i * 2;
    float4 a = s[0];
    float4 b = s[1];
    union { __bf16 h[8]; uint4 q; } u;
    u.h[0] = (__bf16)a.x; u.h[1] = (__bf16)a.y;
    u.h[2] = (__bf16)a.z; u.h[3] = (__bf16)a.w;
    u.h[4] = (__bf16)b.x; u.h[5] = (__bf16)b.y;
    u.h[6] = (__bf16)b.z; u.h[7] = (__bf16)b.w;
    *((uint4*)dst + i) = u.q;
  }
}

// ---------------------------------------------------------------------------
// C[m,n] = scale * sum_k A[m,k] * B[n,k]  (x @ W^T, both row-major bf16)
// Block tile 128x128, K-step 32. 8 waves: wave owns 32x64 = 2x4 WMMA tiles.
// OUT_T=true stores C transposed as C[n*M + m] (used for V -> [D, tokens]).
// ---------------------------------------------------------------------------
template <typename TOUT, bool OUT_T>
__global__ __launch_bounds__(256) void gemm_nt_bf16(
    const __bf16* __restrict__ A,   // [M,K]
    const __bf16* __restrict__ Bw,  // [N,K]
    TOUT* __restrict__ C,
    int M, int N, int K, float scale) {
  __shared__ __align__(16) __bf16 lA[128 * 32];
  __shared__ __align__(16) __bf16 lB[128 * 32];

  const int tid   = threadIdx.x;
  const int lane  = tid & 31;
  const int wid   = tid >> 5;
  const int l16   = lane & 15;
  const int lhalf = lane >> 4;
  const int klo   = lhalf * 8;

  const int mBase = blockIdx.y * 128;
  const int nBase = blockIdx.x * 128;
  const int waveM = (wid >> 1) * 32;  // 4 m-waves
  const int waveN = (wid & 1) * 64;   // 2 n-waves

  v8f acc[2][4];
#pragma unroll
  for (int i = 0; i < 2; ++i)
#pragma unroll
    for (int j = 0; j < 4; ++j) acc[i][j] = v8f{};

  union F { v16bf v; uint4 q[2]; };

  for (int k0 = 0; k0 < K; k0 += 32) {
    // Stage 128x32 tiles of A and B into LDS (each thread: 2x16B per tile).
#pragma unroll
    for (int i = 0; i < 2; ++i) {
      int c   = tid + 256 * i;        // chunk id, 512 chunks of 8 elements
      int row = c >> 2;
      int col = (c & 3) * 8;
      stage16(&A[(size_t)(mBase + row) * K + k0 + col], &lA[row * 32 + col]);
      stage16(&Bw[(size_t)(nBase + row) * K + k0 + col], &lB[row * 32 + col]);
    }
    stage_wait();
    __syncthreads();

    if (k0 + 32 < K) {  // hint next tiles into cache (global_prefetch_b8)
      __builtin_prefetch(&A[(size_t)(mBase + (tid >> 1)) * K + k0 + 32], 0, 1);
      __builtin_prefetch(&Bw[(size_t)(nBase + (tid >> 1)) * K + k0 + 32], 0, 1);
    }

    // A fragments: lane=row(M), K split lo/hi by lane-half (ISA 7.12.2)
    F af[2], bfm[4];
#pragma unroll
    for (int ms = 0; ms < 2; ++ms) {
      int row = waveM + ms * 16 + l16;
      af[ms].q[0] = *(const uint4*)&lA[row * 32 + klo];
      af[ms].q[1] = *(const uint4*)&lA[row * 32 + 16 + klo];
    }
#pragma unroll
    for (int ns = 0; ns < 4; ++ns) {
      int row = waveN + ns * 16 + l16;   // B fragment: lane = output column n
      bfm[ns].q[0] = *(const uint4*)&lB[row * 32 + klo];
      bfm[ns].q[1] = *(const uint4*)&lB[row * 32 + 16 + klo];
    }
#pragma unroll
    for (int ms = 0; ms < 2; ++ms)
#pragma unroll
      for (int ns = 0; ns < 4; ++ns)
        acc[ms][ns] = __builtin_amdgcn_wmma_f32_16x16x32_bf16(
            false, af[ms].v, false, bfm[ns].v, (short)0, acc[ms][ns],
            false, false);
    __syncthreads();
  }

  // Epilogue: C layout is lane=column, VGPR r = row (r<8 -> M+r / M+8+r).
#pragma unroll
  for (int ms = 0; ms < 2; ++ms)
#pragma unroll
    for (int ns = 0; ns < 4; ++ns)
#pragma unroll
      for (int r = 0; r < 8; ++r) {
        int m = mBase + waveM + ms * 16 + lhalf * 8 + r;
        int n = nBase + waveN + ns * 16 + l16;
        float v = acc[ms][ns][r] * scale;
        if (OUT_T)
          C[(size_t)n * M + m] = (TOUT)v;
        else
          C[(size_t)m * N + n] = (TOUT)v;
      }
}

// ---------------------------------------------------------------------------
// Flash MQA. Each wave: 16 query rows x full head dim (128), keys streamed
// in chunks of 64. Q is pre-scaled by (1/sqrt(D))*log2(e), softmax runs in
// exp2 domain. Row-max reduced with shfl_xor across the 16 lanes holding a
// row (C layout). The softmax denominator is accumulated by an extra WMMA
// against an all-ones B matrix (rowsum broadcast across lanes), so no
// cross-lane sum reduction is needed.
// Q: [B,S,HIDDEN] bf16 ; Kc: [B,S,128] bf16 ; Vt: [128, NTOK] bf16 (V^T)
// ---------------------------------------------------------------------------
__global__ __launch_bounds__(256) void flash_mqa(
    const __bf16* __restrict__ Q, const __bf16* __restrict__ Kc,
    const __bf16* __restrict__ Vt, __bf16* __restrict__ O) {
  __shared__ __align__(16) __bf16 pScratch[8][16 * 64];

  const int lane  = threadIdx.x & 31;
  const int wid   = threadIdx.x >> 5;
  const int l16   = lane & 15;
  const int lhalf = lane >> 4;
  const int klo   = lhalf * 8;

  const int h   = blockIdx.y;
  const int b   = blockIdx.z;
  const int qm0 = blockIdx.x * 128 + wid * 16;  // first query row of this wave

  union F { v16bf v; uint4 q[2]; };

  // all-ones B fragment (32x16 matrix of 1.0 -> every element 1 in any layout)
  v16bf ones;
#pragma unroll
  for (int i = 0; i < 16; ++i) ones[i] = (__bf16)1.0f;

  // Q fragments: 4 K-chunks of 32 over the 128-wide head dim (pre-scaled).
  F qf[4];
  {
    const __bf16* qrow =
        Q + ((size_t)(b * SEQ + qm0 + l16)) * HIDDEN + h * HEAD_DIM;
#pragma unroll
    for (int c = 0; c < 4; ++c) {
      qf[c].q[0] = *(const uint4*)&qrow[c * 32 + klo];
      qf[c].q[1] = *(const uint4*)&qrow[c * 32 + 16 + klo];
    }
  }

  v8f o_acc[8];
#pragma unroll
  for (int i = 0; i < 8; ++i) o_acc[i] = v8f{};
  v8f l_acc = v8f{};
  float mrow[8];
#pragma unroll
  for (int r = 0; r < 8; ++r) mrow[r] = -3.0e38f;

  for (int kc = 0; kc < SEQ; kc += 64) {
    // ---- S = Q @ K^T over this 64-key chunk (four 16-column tiles) ----
    v8f s[4];
#pragma unroll
    for (int nt = 0; nt < 4; ++nt) {
      s[nt] = v8f{};
      const __bf16* krow =
          Kc + ((size_t)(b * SEQ + kc + nt * 16 + l16)) * HEAD_DIM;
#pragma unroll
      for (int c = 0; c < 4; ++c) {
        F kf;
        kf.q[0] = *(const uint4*)&krow[c * 32 + klo];
        kf.q[1] = *(const uint4*)&krow[c * 32 + 16 + klo];
        s[nt] = __builtin_amdgcn_wmma_f32_16x16x32_bf16(
            false, qf[c].v, false, kf.v, (short)0, s[nt], false, false);
      }
    }

    // ---- online softmax (exp2 domain); row r = component r across lanes ----
    float alpha[8];
#pragma unroll
    for (int r = 0; r < 8; ++r) {
      float mx = fmaxf(fmaxf(s[0][r], s[1][r]), fmaxf(s[2][r], s[3][r]));
#pragma unroll
      for (int d = 1; d < 16; d <<= 1) mx = fmaxf(mx, __shfl_xor(mx, d, 32));
      float mnew = fmaxf(mrow[r], mx);
      alpha[r]   = fast_exp2(mrow[r] - mnew);
      mrow[r]    = mnew;
#pragma unroll
      for (int nt = 0; nt < 4; ++nt) s[nt][r] = fast_exp2(s[nt][r] - mnew);
    }
#pragma unroll
    for (int i = 0; i < 8; ++i)
#pragma unroll
      for (int r = 0; r < 8; ++r) o_acc[i][r] *= alpha[r];
#pragma unroll
    for (int r = 0; r < 8; ++r) l_acc[r] *= alpha[r];

    // ---- P: C layout -> A layout via per-wave LDS scratch ----
    __bf16* ps = &pScratch[wid][0];
#pragma unroll
    for (int nt = 0; nt < 4; ++nt)
#pragma unroll
      for (int r = 0; r < 8; ++r)
        ps[(r + 8 * lhalf) * 64 + nt * 16 + l16] = (__bf16)s[nt][r];
    asm volatile("" ::: "memory");  // keep same-wave DS stores before loads
    F pf[2];
    pf[0].q[0] = *(const uint4*)&ps[l16 * 64 + klo];
    pf[0].q[1] = *(const uint4*)&ps[l16 * 64 + 16 + klo];
    pf[1].q[0] = *(const uint4*)&ps[l16 * 64 + 32 + klo];
    pf[1].q[1] = *(const uint4*)&ps[l16 * 64 + 48 + klo];

    // ---- denominator: l += rowsum(P) via ones-column WMMA ----
    l_acc = __builtin_amdgcn_wmma_f32_16x16x32_bf16(
        false, pf[0].v, false, ones, (short)0, l_acc, false, false);
    l_acc = __builtin_amdgcn_wmma_f32_16x16x32_bf16(
        false, pf[1].v, false, ones, (short)0, l_acc, false, false);

    // ---- O += P @ Vchunk : 8 dim-tiles x 2 key-subchunks (K=32) ----
#pragma unroll
    for (int ntile = 0; ntile < 8; ++ntile) {
      const __bf16* vrow =
          Vt + (size_t)(ntile * 16 + l16) * NTOK + b * SEQ + kc;
#pragma unroll
      for (int kk = 0; kk < 2; ++kk) {
        F vf;
        vf.q[0] = *(const uint4*)&vrow[kk * 32 + klo];
        vf.q[1] = *(const uint4*)&vrow[kk * 32 + 16 + klo];
        o_acc[ntile] = __builtin_amdgcn_wmma_f32_16x16x32_bf16(
            false, pf[kk].v, false, vf.v, (short)0, o_acc[ntile], false, false);
      }
    }
  }

  // ---- normalize and store attention output (bf16, [B,S,HIDDEN]) ----
  float inv[8];
#pragma unroll
  for (int r = 0; r < 8; ++r) inv[r] = fast_rcp(l_acc[r]);
#pragma unroll
  for (int ntile = 0; ntile < 8; ++ntile)
#pragma unroll
    for (int r = 0; r < 8; ++r) {
      int row = qm0 + lhalf * 8 + r;
      int dim = ntile * 16 + l16;
      O[((size_t)(b * SEQ + row)) * HIDDEN + h * HEAD_DIM + dim] =
          (__bf16)(o_acc[ntile][r] * inv[r]);
    }
}

// ---------------------------------------------------------------------------
// Host-side orchestration
// ---------------------------------------------------------------------------
extern "C" void kernel_launch(void* const* d_in, const int* in_sizes, int n_in,
                              void* d_out, int out_size, void* d_ws,
                              size_t ws_size, hipStream_t stream) {
  const float* x  = (const float*)d_in[0];
  const float* qw = (const float*)d_in[1];
  const float* kw = (const float*)d_in[2];
  const float* vw = (const float*)d_in[3];
  const float* ow = (const float*)d_in[4];
  float* out = (float*)d_out;

  char* ws = (char*)d_ws;
  size_t off = 0;
  auto alloc = [&](size_t elems) {
    void* p = ws + off;
    off += elems * sizeof(__bf16);
    off = (off + 255) & ~(size_t)255;
    return (__bf16*)p;
  };
  __bf16* xb  = alloc((size_t)NTOK * HIDDEN);
  __bf16* qwb = alloc((size_t)HIDDEN * HIDDEN);
  __bf16* kwb = alloc((size_t)HEAD_DIM * HIDDEN);
  __bf16* vwb = alloc((size_t)HEAD_DIM * HIDDEN);
  __bf16* owb = alloc((size_t)HIDDEN * HIDDEN);
  __bf16* qb  = alloc((size_t)NTOK * HIDDEN);
  __bf16* kb  = alloc((size_t)NTOK * HEAD_DIM);
  __bf16* vt  = alloc((size_t)HEAD_DIM * NTOK);  // V transposed [D, tokens]
  __bf16* ab  = alloc((size_t)NTOK * HIDDEN);
  (void)ws_size; (void)n_in; (void)in_sizes; (void)out_size;

  auto cvt = [&](const float* s, __bf16* d, size_t n) {
    int n8 = (int)(n / 8);
    int blocks = (n8 + 255) / 256;
    if (blocks > 2048) blocks = 2048;
    cvt_f32_bf16<<<dim3(blocks), dim3(256), 0, stream>>>(s, d, n8);
  };
  cvt(x,  xb,  (size_t)NTOK * HIDDEN);
  cvt(qw, qwb, (size_t)HIDDEN * HIDDEN);
  cvt(kw, kwb, (size_t)HEAD_DIM * HIDDEN);
  cvt(vw, vwb, (size_t)HEAD_DIM * HIDDEN);
  cvt(ow, owb, (size_t)HIDDEN * HIDDEN);

  // Q = (x @ qw^T) * (1/sqrt(D))*log2(e) -> bf16 [NTOK, HIDDEN]
  gemm_nt_bf16<__bf16, false><<<dim3(HIDDEN / 128, NTOK / 128), 256, 0,
                                stream>>>(xb, qwb, qb, NTOK, HIDDEN, HIDDEN,
                                          ATTN_SCALE * LOG2E);
  // K = x @ kw^T -> bf16 [NTOK, 128]
  gemm_nt_bf16<__bf16, false><<<dim3(1, NTOK / 128), 256, 0, stream>>>(
      xb, kwb, kb, NTOK, HEAD_DIM, HIDDEN, 1.0f);
  // V^T = (x @ vw^T)^T -> bf16 [128, NTOK]
  gemm_nt_bf16<__bf16, true><<<dim3(1, NTOK / 128), 256, 0, stream>>>(
      xb, vwb, vt, NTOK, HEAD_DIM, HIDDEN, 1.0f);

  // Flash attention: grid (S/128 q-tiles, heads, batch)
  flash_mqa<<<dim3(SEQ / 128, HEADS, BATCH), 256, 0, stream>>>(qb, kb, vt, ab);

  // out = attn_out @ ow^T -> fp32 [NTOK, HIDDEN]
  gemm_nt_bf16<float, false><<<dim3(HIDDEN / 128, NTOK / 128), 256, 0,
                               stream>>>(ab, owb, out, NTOK, HIDDEN, HIDDEN,
                                         1.0f);
}
